// TemporalBlock_16518444221144
// MI455X (gfx1250) — matrix operand order repaired and checked by reading, not compiled, not run on hardware
//
#include <hip/hip_runtime.h>
#include <hip/hip_bf16.h>
#include <math.h>

// ---------------------------------------------------------------------------
// Problem constants (match reference)
// ---------------------------------------------------------------------------
#define BATCH 4
#define SEQ   1024
#define DMODEL 512
#define NHEAD 8
#define HDIM  64
#define FFDIM 2048
#define EPS   1e-5f

typedef __attribute__((ext_vector_type(16))) _Float16 v16h;
typedef __attribute__((ext_vector_type(8)))  _Float16 v8h;
typedef __attribute__((ext_vector_type(8)))  float    v8f;

#define BK 32
#define LDSS 40   // LDS row stride in halves (32 + 8 pad, keeps 16B alignment)

// ---------------------------------------------------------------------------
// WMMA fragment loaders (CDNA5 16x16x32 f16 layouts, ISA 7.12.2)
// A 16x32: lane L -> row M = L&15, half h = L>>4;
//   elems 0..7  -> K = h*8 + (0..7)
//   elems 8..15 -> K = 16 + h*8 + (0..7)
// B 32x16 (stored transposed in LDS as [N][K]): lane L -> col N = L&15,
//   elems 0..7 -> K = (L>>4)*8 + e; elems 8..15 -> K = 16 + (L>>4)*8 + (e-8)
// ---------------------------------------------------------------------------
__device__ __forceinline__ v16h frag_a(const _Float16* sA, int mrow, int lane) {
  const int m  = mrow + (lane & 15);
  const int hh = (lane >> 4) * 8;
  const _Float16* p = sA + m * LDSS + hh;
  v8h lo = *(const v8h*)(p);
  v8h hi = *(const v8h*)(p + 16);
  v16h r;
#pragma unroll
  for (int i = 0; i < 8; ++i) { r[i] = lo[i]; r[i + 8] = hi[i]; }
  return r;
}

__device__ __forceinline__ v16h frag_b(const _Float16* sBt, int ncol, int lane) {
  const int n  = ncol + (lane & 15);
  const int kb = (lane >> 4) * 8;
  const _Float16* p = sBt + n * LDSS + kb;
  v8h lo = *(const v8h*)(p);
  v8h hi = *(const v8h*)(p + 16);
  v16h r;
#pragma unroll
  for (int i = 0; i < 8; ++i) { r[i] = lo[i]; r[i + 8] = hi[i]; }
  return r;
}

__device__ __forceinline__ v8f wmma_f16(v16h a, v16h b, v8f c) {
  return __builtin_amdgcn_wmma_f32_16x16x32_f16(false, a, false, b,
                                                (short)0, c, false, false);
}

// ---------------------------------------------------------------------------
// Generic GEMM:  C[M,N] = act(A[M,K] @ B[K,N] + bias[N])
// 256 threads (8 waves), 128x128 block tile, BK=32.
// Wave (wm, wn): 32x64 patch -> 2 A frags x 4 B frags = 8 WMMA per K-step.
// ACT: 0 = none, 1 = ELU  (compile-time, no per-element flag tests)
// All in-block addressing is 32-bit (offsets < 2^31 bytes).
// ---------------------------------------------------------------------------
template <int ACT>
__global__ __launch_bounds__(256)
void gemm_bias_act(const float* __restrict__ A, const float* __restrict__ Bw,
                   const float* __restrict__ bias, float* __restrict__ C,
                   int K, int lda, int ldb, int ldc) {
  __shared__ __align__(16) _Float16 sA[128 * LDSS];
  __shared__ __align__(16) _Float16 sB[128 * LDSS];   // transposed: [n][k]
  const int tid  = threadIdx.x;
  const int lane = tid & 31;
  const int wave = tid >> 5;
  const int m0 = blockIdx.y * 128;
  const int n0 = blockIdx.x * 128;
  const int wm = (wave >> 1) * 32;
  const int wn = (wave & 1) * 64;

  v8f acc[2][4];
#pragma unroll
  for (int i = 0; i < 2; ++i)
#pragma unroll
    for (int j = 0; j < 4; ++j)
#pragma unroll
      for (int r = 0; r < 8; ++r) acc[i][j][r] = 0.0f;

  for (int k0 = 0; k0 < K; k0 += BK) {
    // Prefetch next K-step tiles (global_prefetch_b8 path)
    if (k0 + BK < K) {
      __builtin_prefetch(A + (m0 + (tid >> 3)) * lda + (k0 + BK) + ((tid & 7) << 2), 0, 1);
      __builtin_prefetch(Bw + (k0 + BK + (tid >> 5)) * ldb + n0 + ((tid & 31) << 2), 0, 1);
    }
    // A tile 128x32 f32 -> f16 LDS (1024 float4 loads over 256 threads)
#pragma unroll
    for (int i = 0; i < 4; ++i) {
      const int lin = tid + i * 256;
      const int r   = lin >> 3;
      const int c4  = (lin & 7) << 2;
      const float4 vv = *(const float4*)(A + (m0 + r) * lda + k0 + c4);
      _Float16* dst = sA + r * LDSS + c4;
      dst[0] = (_Float16)vv.x; dst[1] = (_Float16)vv.y;
      dst[2] = (_Float16)vv.z; dst[3] = (_Float16)vv.w;
    }
    // B tile 32x128 -> transposed f16 LDS
#pragma unroll
    for (int i = 0; i < 4; ++i) {
      const int lin = tid + i * 256;
      const int r   = lin >> 5;          // k row
      const int c4  = (lin & 31) << 2;   // n col
      const float4 vv = *(const float4*)(Bw + (k0 + r) * ldb + n0 + c4);
      sB[(c4 + 0) * LDSS + r] = (_Float16)vv.x;
      sB[(c4 + 1) * LDSS + r] = (_Float16)vv.y;
      sB[(c4 + 2) * LDSS + r] = (_Float16)vv.z;
      sB[(c4 + 3) * LDSS + r] = (_Float16)vv.w;
    }
    __syncthreads();
    v16h af0 = frag_a(sA, wm, lane);
    v16h af1 = frag_a(sA, wm + 16, lane);
    v16h bf0 = frag_b(sB, wn +  0, lane);
    v16h bf1 = frag_b(sB, wn + 16, lane);
    v16h bf2 = frag_b(sB, wn + 32, lane);
    v16h bf3 = frag_b(sB, wn + 48, lane);
    acc[0][0] = wmma_f16(af0, bf0, acc[0][0]);
    acc[0][1] = wmma_f16(af0, bf1, acc[0][1]);
    acc[0][2] = wmma_f16(af0, bf2, acc[0][2]);
    acc[0][3] = wmma_f16(af0, bf3, acc[0][3]);
    acc[1][0] = wmma_f16(af1, bf0, acc[1][0]);
    acc[1][1] = wmma_f16(af1, bf1, acc[1][1]);
    acc[1][2] = wmma_f16(af1, bf2, acc[1][2]);
    acc[1][3] = wmma_f16(af1, bf3, acc[1][3]);
    __syncthreads();
  }

  const int nlane = lane & 15;
  const int mhalf = (lane >> 4) * 8;
#pragma unroll
  for (int j = 0; j < 4; ++j) {
    const int col = n0 + wn + j * 16 + nlane;
    const float bvv = bias[col];
#pragma unroll
    for (int i = 0; i < 2; ++i) {
      const int row = m0 + wm + i * 16 + mhalf;
      float* cp = C + row * ldc + col;
#pragma unroll
      for (int r = 0; r < 8; ++r) {
        float val = acc[i][j][r] + bvv;
        if (ACT == 1) val = val > 0.0f ? val : (expf(val) - 1.0f);
        cp[r * ldc] = val;
      }
    }
  }
}

// ---------------------------------------------------------------------------
// scores[bh, q, k] = (q_vec . k_vec) / 8   -- tiles fully above diagonal skipped
// grid (S/64, S/64, B*H), 128 threads (4 waves), 64x64 tile
// ---------------------------------------------------------------------------
__global__ __launch_bounds__(128)
void attn_scores(const float* __restrict__ qm, const float* __restrict__ km,
                 float* __restrict__ attn) {
  const int k0t = blockIdx.x * 64;
  const int q0t = blockIdx.y * 64;
  if (k0t > q0t + 63) return;                       // fully causal-masked tile
  const int bh = blockIdx.z;
  const int b = bh >> 3, h = bh & 7;
  const float* qb = qm + (size_t)b * SEQ * DMODEL + h * HDIM;
  const float* kb = km + (size_t)b * SEQ * DMODEL + h * HDIM;
  float* ob = attn + (size_t)bh * SEQ * SEQ;

  __shared__ __align__(16) _Float16 sA[64 * LDSS];
  __shared__ __align__(16) _Float16 sB[64 * LDSS];
  const int tid  = threadIdx.x;
  const int lane = tid & 31;
  const int wave = tid >> 5;
  const int wm = (wave >> 1) * 32;
  const int wn = (wave & 1) * 32;

  v8f acc[2][2];
#pragma unroll
  for (int i = 0; i < 2; ++i)
#pragma unroll
    for (int j = 0; j < 2; ++j)
#pragma unroll
      for (int r = 0; r < 8; ++r) acc[i][j][r] = 0.0f;

  for (int kc = 0; kc < HDIM; kc += BK) {
#pragma unroll
    for (int i = 0; i < 4; ++i) {                   // A: q rows (64x32)
      const int lin = tid + i * 128;
      const int r = lin >> 3, c4 = (lin & 7) << 2;
      const float4 vv = *(const float4*)(qb + (q0t + r) * DMODEL + kc + c4);
      _Float16* dst = sA + r * LDSS + c4;
      dst[0] = (_Float16)vv.x; dst[1] = (_Float16)vv.y;
      dst[2] = (_Float16)vv.z; dst[3] = (_Float16)vv.w;
    }
#pragma unroll
    for (int i = 0; i < 4; ++i) {                   // B^T = k rows (64x32) as-is
      const int lin = tid + i * 128;
      const int r = lin >> 3, c4 = (lin & 7) << 2;
      const float4 vv = *(const float4*)(kb + (k0t + r) * DMODEL + kc + c4);
      _Float16* dst = sB + r * LDSS + c4;
      dst[0] = (_Float16)vv.x; dst[1] = (_Float16)vv.y;
      dst[2] = (_Float16)vv.z; dst[3] = (_Float16)vv.w;
    }
    __syncthreads();
    v16h af0 = frag_a(sA, wm, lane);
    v16h af1 = frag_a(sA, wm + 16, lane);
    v16h bf0 = frag_b(sB, wn, lane);
    v16h bf1 = frag_b(sB, wn + 16, lane);
    acc[0][0] = wmma_f16(af0, bf0, acc[0][0]);
    acc[0][1] = wmma_f16(af0, bf1, acc[0][1]);
    acc[1][0] = wmma_f16(af1, bf0, acc[1][0]);
    acc[1][1] = wmma_f16(af1, bf1, acc[1][1]);
    __syncthreads();
  }

  const int nlane = lane & 15;
  const int mhalf = (lane >> 4) * 8;
#pragma unroll
  for (int i = 0; i < 2; ++i)
#pragma unroll
    for (int j = 0; j < 2; ++j) {
      const int col = k0t + wn + j * 16 + nlane;
      const int row = q0t + wm + i * 16 + mhalf;
      float* cp = ob + row * SEQ + col;
#pragma unroll
      for (int r = 0; r < 8; ++r)
        cp[r * SEQ] = acc[i][j][r] * 0.125f;        // 1/sqrt(64)
    }
}

// ---------------------------------------------------------------------------
// Reductions (wave32)
// ---------------------------------------------------------------------------
__device__ __forceinline__ float wave_sum(float v) {
#pragma unroll
  for (int o = 16; o > 0; o >>= 1) v += __shfl_down(v, o, 32);
  return v;
}
__device__ __forceinline__ float wave_max(float v) {
#pragma unroll
  for (int o = 16; o > 0; o >>= 1) v = fmaxf(v, __shfl_down(v, o, 32));
  return v;
}
__device__ __forceinline__ float block_sum(float v, float* sm) {
  const int lane = threadIdx.x & 31, w = threadIdx.x >> 5;
  const int nw = blockDim.x >> 5;
  v = wave_sum(v);
  if (lane == 0) sm[w] = v;
  __syncthreads();
  float r = (threadIdx.x < nw) ? sm[threadIdx.x] : 0.0f;
  if (w == 0) { r = wave_sum(r); if (lane == 0) sm[0] = r; }
  __syncthreads();
  const float res = sm[0];
  __syncthreads();
  return res;
}
__device__ __forceinline__ float block_max(float v, float* sm) {
  const int lane = threadIdx.x & 31, w = threadIdx.x >> 5;
  const int nw = blockDim.x >> 5;
  v = wave_max(v);
  if (lane == 0) sm[w] = v;
  __syncthreads();
  float r = (threadIdx.x < nw) ? sm[threadIdx.x] : -INFINITY;
  if (w == 0) { r = wave_max(r); if (lane == 0) sm[0] = r; }
  __syncthreads();
  const float res = sm[0];
  __syncthreads();
  return res;
}

// ---------------------------------------------------------------------------
// In-place causal softmax over one attention row; masked tail written as 0.
// grid (S, B*H), 256 threads
// ---------------------------------------------------------------------------
__global__ __launch_bounds__(256)
void softmax_causal(float* __restrict__ attn) {
  const int qrow = blockIdx.x;
  const int bh   = blockIdx.y;
  float* row = attn + (size_t)bh * SEQ * SEQ + (size_t)qrow * SEQ;
  const int L = qrow + 1;
  __shared__ float red[8];
  const int tid = threadIdx.x;

  float vals[4];
  float vmax = -INFINITY;
#pragma unroll
  for (int i = 0; i < 4; ++i) {
    const int j = tid + i * 256;
    const float x = (j < L) ? row[j] : -INFINITY;
    vals[i] = x;
    vmax = fmaxf(vmax, x);
  }
  vmax = block_max(vmax, red);

  float vsum = 0.0f;
#pragma unroll
  for (int i = 0; i < 4; ++i) {
    const int j = tid + i * 256;
    const float e = (j < L) ? expf(vals[i] - vmax) : 0.0f;
    vals[i] = e;
    vsum += e;
  }
  vsum = block_sum(vsum, red);
  const float inv = 1.0f / vsum;
#pragma unroll
  for (int i = 0; i < 4; ++i) {
    const int j = tid + i * 256;
    row[j] = vals[i] * inv;          // zeros in the masked region
  }
}

// ---------------------------------------------------------------------------
// attn_out[b, q, h*64+n] = sum_k attn[bh,q,k] * v[b,k,h*64+n]
// Causal: K loop stops at the diagonal tile (weights beyond are exact zeros).
// grid (S/64, B*H), 128 threads
// ---------------------------------------------------------------------------
__global__ __launch_bounds__(128)
void attn_av(const float* __restrict__ attn, const float* __restrict__ vm,
             float* __restrict__ o) {
  const int qt = blockIdx.x;
  const int bh = blockIdx.y;
  const int b = bh >> 3, h = bh & 7;
  const float* ab = attn + (size_t)bh * SEQ * SEQ + (size_t)qt * 64 * SEQ;
  const float* vb = vm + (size_t)b * SEQ * DMODEL + h * HDIM;
  float* ob = o + (size_t)b * SEQ * DMODEL + (size_t)qt * 64 * DMODEL + h * HDIM;

  __shared__ __align__(16) _Float16 sA[64 * LDSS];
  __shared__ __align__(16) _Float16 sB[64 * LDSS];
  const int tid  = threadIdx.x;
  const int lane = tid & 31;
  const int wave = tid >> 5;
  const int wm = (wave >> 1) * 32;
  const int wn = (wave & 1) * 32;

  v8f acc[2][2];
#pragma unroll
  for (int i = 0; i < 2; ++i)
#pragma unroll
    for (int j = 0; j < 2; ++j)
#pragma unroll
      for (int r = 0; r < 8; ++r) acc[i][j][r] = 0.0f;

  const int kmax = (qt + 1) * 64;
  for (int k0 = 0; k0 < kmax; k0 += BK) {
#pragma unroll
    for (int i = 0; i < 4; ++i) {                   // A: attn rows 64x32
      const int lin = tid + i * 128;
      const int r = lin >> 3, c4 = (lin & 7) << 2;
      const float4 vv = *(const float4*)(ab + r * SEQ + k0 + c4);
      _Float16* dst = sA + r * LDSS + c4;
      dst[0] = (_Float16)vv.x; dst[1] = (_Float16)vv.y;
      dst[2] = (_Float16)vv.z; dst[3] = (_Float16)vv.w;
    }
#pragma unroll
    for (int i = 0; i < 4; ++i) {                   // B: v rows 32x64 -> [n][k]
      const int lin = tid + i * 128;
      const int r = lin >> 4, c4 = (lin & 15) << 2;
      const float4 vv = *(const float4*)(vb + (k0 + r) * DMODEL + c4);
      sB[(c4 + 0) * LDSS + r] = (_Float16)vv.x;
      sB[(c4 + 1) * LDSS + r] = (_Float16)vv.y;
      sB[(c4 + 2) * LDSS + r] = (_Float16)vv.z;
      sB[(c4 + 3) * LDSS + r] = (_Float16)vv.w;
    }
    __syncthreads();
    v16h af0 = frag_a(sA, wm, lane);
    v16h af1 = frag_a(sA, wm + 16, lane);
    v16h bf0 = frag_b(sB, wn, lane);
    v16h bf1 = frag_b(sB, wn + 16, lane);
    acc[0][0] = wmma_f16(af0, bf0, acc[0][0]);
    acc[0][1] = wmma_f16(af0, bf1, acc[0][1]);
    acc[1][0] = wmma_f16(af1, bf0, acc[1][0]);
    acc[1][1] = wmma_f16(af1, bf1, acc[1][1]);
    __syncthreads();
  }

  const int nlane = lane & 15;
  const int mhalf = (lane >> 4) * 8;
#pragma unroll
  for (int i = 0; i < 2; ++i)
#pragma unroll
    for (int j = 0; j < 2; ++j) {
      const int col = wn + j * 16 + nlane;
      const int row = wm + i * 16 + mhalf;
      float* cp = ob + row * DMODEL + col;
#pragma unroll
      for (int r = 0; r < 8; ++r)
        cp[r * DMODEL] = acc[i][j][r];
    }
}

// ---------------------------------------------------------------------------
// out = LN(x + y) * g + b     (row of 512, 256 threads x 2)
// ---------------------------------------------------------------------------
__global__ __launch_bounds__(256)
void add_layernorm(const float* __restrict__ x, const float* __restrict__ y,
                   const float* __restrict__ gw, const float* __restrict__ bw,
                   float* __restrict__ out) {
  const size_t row = blockIdx.x;
  const int tid = threadIdx.x;
  __shared__ float red[8];
  const float* xr = x + row * DMODEL;
  const float* yr = y + row * DMODEL;
  const float v0 = xr[tid] + yr[tid];
  const float v1 = xr[tid + 256] + yr[tid + 256];
  const float mu = block_sum(v0 + v1, red) * (1.0f / DMODEL);
  const float d0 = v0 - mu, d1 = v1 - mu;
  const float var = block_sum(d0 * d0 + d1 * d1, red) * (1.0f / DMODEL);
  const float rs = rsqrtf(var + EPS);
  out[row * DMODEL + tid]       = d0 * rs * gw[tid]       + bw[tid];
  out[row * DMODEL + tid + 256] = d1 * rs * gw[tid + 256] + bw[tid + 256];
}

// ---------------------------------------------------------------------------
// glu = g[:512]*sigmoid(g[512:]); out = LN(glu + h)*g2 + b2 + h
// ---------------------------------------------------------------------------
__global__ __launch_bounds__(256)
void glu_ln_residual(const float* __restrict__ g, const float* __restrict__ h,
                     const float* __restrict__ gw, const float* __restrict__ bw,
                     float* __restrict__ out) {
  const size_t row = blockIdx.x;
  const int tid = threadIdx.x;
  __shared__ float red[8];
  const float* gr = g + row * (2 * DMODEL);
  const float* hr = h + row * DMODEL;
  const float h0 = hr[tid], h1 = hr[tid + 256];
  const float s0 = 1.0f / (1.0f + expf(-gr[DMODEL + tid]));
  const float s1 = 1.0f / (1.0f + expf(-gr[DMODEL + tid + 256]));
  const float t0 = gr[tid] * s0 + h0;
  const float t1 = gr[tid + 256] * s1 + h1;
  const float mu = block_sum(t0 + t1, red) * (1.0f / DMODEL);
  const float d0 = t0 - mu, d1 = t1 - mu;
  const float var = block_sum(d0 * d0 + d1 * d1, red) * (1.0f / DMODEL);
  const float rs = rsqrtf(var + EPS);
  out[row * DMODEL + tid]       = d0 * rs * gw[tid]       + bw[tid]       + h0;
  out[row * DMODEL + tid + 256] = d1 * rs * gw[tid + 256] + bw[tid + 256] + h1;
}

// ---------------------------------------------------------------------------
// Host-side orchestration
// ---------------------------------------------------------------------------
extern "C" void kernel_launch(void* const* d_in, const int* in_sizes, int n_in,
                              void* d_out, int out_size, void* d_ws, size_t ws_size,
                              hipStream_t stream) {
  const float* x    = (const float*)d_in[0];
  const float* Wq   = (const float*)d_in[1];
  const float* bq   = (const float*)d_in[2];
  const float* Wk   = (const float*)d_in[3];
  const float* bk   = (const float*)d_in[4];
  const float* Wv   = (const float*)d_in[5];
  const float* bv   = (const float*)d_in[6];
  const float* Wo   = (const float*)d_in[7];
  const float* bo   = (const float*)d_in[8];
  const float* ln1g = (const float*)d_in[9];
  const float* ln1b = (const float*)d_in[10];
  const float* W1   = (const float*)d_in[11];
  const float* b1   = (const float*)d_in[12];
  const float* W2   = (const float*)d_in[13];
  const float* b2   = (const float*)d_in[14];
  const float* Wg   = (const float*)d_in[15];
  const float* bg   = (const float*)d_in[16];
  const float* ln2g = (const float*)d_in[17];
  const float* ln2b = (const float*)d_in[18];

  float* out  = (float*)d_out;
  float* attn = out + (size_t)BATCH * SEQ * DMODEL;   // (B,H,S,S) slice of d_out

  const size_t SD = (size_t)BATCH * SEQ * DMODEL;     // 2M floats
  float* ws    = (float*)d_ws;
  float* qb    = ws;             // [0, SD)
  float* kb    = ws + SD;        // [SD, 2SD)
  float* vb    = ws + 2 * SD;    // [2SD, 3SD)
  float* hb    = ws + 3 * SD;    // [3SD, 4SD)  lives till the end
  float* ff1   = ws + 4 * SD;    // [4SD, 8SD)  (B*S*FF)
  float* attno = qb;             // reuse: q dead after scores
  float* aproj = kb;             // reuse: k dead after scores
  float* ff2   = vb;             // reuse: v dead after A*V
  float* gbuf  = ff1;            // reuse: ff1 dead after W2 gemm
  (void)in_sizes; (void)n_in; (void)out_size; (void)ws_size;

  const dim3 blk128(128), blk256(256);
  const int Mrows = BATCH * SEQ;                      // 4096

  // QKV projections (128x128 block tiles)
  gemm_bias_act<0><<<dim3(DMODEL / 128, Mrows / 128), blk256, 0, stream>>>(
      x, Wq, bq, qb, DMODEL, DMODEL, DMODEL, DMODEL);
  gemm_bias_act<0><<<dim3(DMODEL / 128, Mrows / 128), blk256, 0, stream>>>(
      x, Wk, bk, kb, DMODEL, DMODEL, DMODEL, DMODEL);
  gemm_bias_act<0><<<dim3(DMODEL / 128, Mrows / 128), blk256, 0, stream>>>(
      x, Wv, bv, vb, DMODEL, DMODEL, DMODEL, DMODEL);

  // scores -> d_out attn slice (raw, scaled), then in-place causal softmax
  attn_scores<<<dim3(SEQ / 64, SEQ / 64, BATCH * NHEAD), blk128, 0, stream>>>(
      qb, kb, attn);
  softmax_causal<<<dim3(SEQ, BATCH * NHEAD), blk256, 0, stream>>>(attn);

  // attn @ V  (causally truncated K loop)
  attn_av<<<dim3(SEQ / 64, BATCH * NHEAD), blk128, 0, stream>>>(attn, vb, attno);

  // output projection + residual LN
  gemm_bias_act<0><<<dim3(DMODEL / 128, Mrows / 128), blk256, 0, stream>>>(
      attno, Wo, bo, aproj, DMODEL, DMODEL, DMODEL, DMODEL);
  add_layernorm<<<dim3(Mrows), blk256, 0, stream>>>(x, aproj, ln1g, ln1b, hb);

  // GRN: elu(h@W1) @ W2, gate via Wg, GLU + LN + residual
  gemm_bias_act<1><<<dim3(FFDIM / 128, Mrows / 128), blk256, 0, stream>>>(
      hb, W1, b1, ff1, DMODEL, DMODEL, FFDIM, FFDIM);
  gemm_bias_act<0><<<dim3(DMODEL / 128, Mrows / 128), blk256, 0, stream>>>(
      ff1, W2, b2, ff2, FFDIM, FFDIM, DMODEL, DMODEL);
  gemm_bias_act<0><<<dim3((2 * DMODEL) / 128, Mrows / 128), blk256, 0, stream>>>(
      ff2, Wg, bg, gbuf, DMODEL, DMODEL, 2 * DMODEL, 2 * DMODEL);
  glu_ln_residual<<<dim3(Mrows), blk256, 0, stream>>>(gbuf, hb, ln2g, ln2b, out);
}
